// MoEClassical_81887846465603
// MI455X (gfx1250) — matrix operand order repaired and checked by reading, not compiled, not run on hardware
//
#include <hip/hip_runtime.h>
#include <stdint.h>

#define T_TOK 4096
#define H_DIM 1024
#define E_NUM 8
#define FF_DIM 4096

typedef __attribute__((ext_vector_type(16))) __bf16 v16bf;
typedef __attribute__((ext_vector_type(8)))  float  v8f;
typedef __attribute__((ext_vector_type(4)))  unsigned int u32x4;
typedef __attribute__((ext_vector_type(2)))  unsigned int u32x2;

union Frag16 { v16bf v; u32x4 q[2]; unsigned short s[16]; };

// native conversion -> v_cvt_pk_bf16_f32
__device__ __forceinline__ unsigned short f2bf(float f) {
    union { __bf16 b; unsigned short u; } c; c.b = (__bf16)f; return c.u;
}

__device__ __forceinline__ void atomic_add_f32(float* p, float v) {
    asm volatile("global_atomic_add_f32 %0, %1, off" :: "v"(p), "v"(v) : "memory");
}
// CDNA5 async LDS<->global DMA (ASYNCcnt-tracked)
__device__ __forceinline__ void async_load_b128(unsigned lds_off, const void* gaddr) {
    asm volatile("global_load_async_to_lds_b128 %0, %1, off" :: "v"(lds_off), "v"(gaddr) : "memory");
}
__device__ __forceinline__ void wait_async0() {
    asm volatile("s_wait_asynccnt 0" ::: "memory");
}

// ---------------- x fp32 -> bf16 once -----------------------------------------
__global__ __launch_bounds__(256) void moe_cvt(const float* __restrict__ x,
                                               unsigned short* __restrict__ xb) {
    const size_t i = ((size_t)blockIdx.x * blockDim.x + threadIdx.x) * 4;
    const float4 v = *(const float4*)(x + i);
    u32x2 o;
    o[0] = (unsigned)f2bf(v.x) | ((unsigned)f2bf(v.y) << 16);
    o[1] = (unsigned)f2bf(v.z) | ((unsigned)f2bf(v.w) << 16);
    *(u32x2*)(xb + i) = o;
}

// ---------------- router: one wave per token ----------------------------------
__global__ __launch_bounds__(256) void moe_router(const float* __restrict__ x,
                                                  const float* __restrict__ Wg,
                                                  int* __restrict__ cnt,
                                                  int* __restrict__ btok,
                                                  float* __restrict__ bw) {
    const int wv = threadIdx.x >> 5;
    const int lane = threadIdx.x & 31;
    const int t = blockIdx.x * 8 + wv;
    const float* xr = x + (size_t)t * H_DIM;

    float a[E_NUM];
    #pragma unroll
    for (int e = 0; e < E_NUM; ++e) a[e] = 0.f;
    for (int i = lane; i < H_DIM; i += 32) {
        const float xv = xr[i];
        const float* wr = Wg + (size_t)i * E_NUM;
        #pragma unroll
        for (int e = 0; e < E_NUM; ++e) a[e] += xv * wr[e];
    }
    #pragma unroll
    for (int off = 16; off > 0; off >>= 1) {
        #pragma unroll
        for (int e = 0; e < E_NUM; ++e) a[e] += __shfl_xor(a[e], off, 32);
    }
    if (lane == 0) {
        float mx = a[0];
        #pragma unroll
        for (int e = 1; e < E_NUM; ++e) mx = fmaxf(mx, a[e]);
        float p[E_NUM];
        #pragma unroll
        for (int e = 0; e < E_NUM; ++e) p[e] = __expf(a[e] - mx);
        int i1 = 0;
        #pragma unroll
        for (int e = 1; e < E_NUM; ++e) if (p[e] > p[i1]) i1 = e;
        int i2 = (i1 == 0) ? 1 : 0;
        #pragma unroll
        for (int e = 0; e < E_NUM; ++e) if (e != i1 && p[e] > p[i2]) i2 = e;
        const float inv = 1.f / (p[i1] + p[i2]);
        int s1 = atomicAdd(&cnt[i1], 1);
        btok[i1 * T_TOK + s1] = t;  bw[i1 * T_TOK + s1] = p[i1] * inv;
        int s2 = atomicAdd(&cnt[i2], 1);
        btok[i2 * T_TOK + s2] = t;  bw[i2 * T_TOK + s2] = p[i2] * inv;
    }
}

// ---------------- exclusive prefix over 8 counts ------------------------------
__global__ void moe_prefix(const int* __restrict__ cnt, int* __restrict__ base) {
    if (threadIdx.x == 0) {
        int s = 0;
        #pragma unroll
        for (int e = 0; e < E_NUM; ++e) { base[e] = s; s += cnt[e]; }
    }
}

// ---------------- GEMM1: h1 = relu(x @ Wfc)^2 -> bf16 workspace ---------------
// Block 512 thr = 16 waves. Tile: 64 tokens x FF-half (2048), chunks of 256.
// A tile (64x1024 bf16 = 129 KB) is LDS-resident: one async DMA prologue, then
// the main loop is barrier-free WMMA streaming.
__global__ __launch_bounds__(512) void moe_fc(const float* __restrict__ Wfc,
                                              const unsigned short* __restrict__ xbf,
                                              const int* __restrict__ cnt,
                                              const int* __restrict__ base,
                                              const int* __restrict__ btok,
                                              unsigned short* __restrict__ h1ws) {
    const int e = blockIdx.y;
    const int ce = cnt[e];
    const int tbase = blockIdx.x * 64;
    if (tbase >= ce) return;
    const int ffbase = blockIdx.z * 2048;

    __shared__ unsigned short Ast[64][1032];   // full x tile (1024 + 8 pad)
    __shared__ int stok[64];

    const int tid   = threadIdx.x;
    const int wv    = tid >> 5;
    const int lane  = tid & 31;
    const int nlane = lane & 15;
    const int kg    = lane >> 4;

    if (tid < 64) {
        const int idx = tbase + tid;
        stok[tid] = (idx < ce) ? btok[e * T_TOK + idx] : -1;
    }
    __syncthreads();

    // prologue: async-DMA the whole 64x1024 bf16 tile into LDS (256 B/thread)
    {
        const int srow = tid >> 3;
        const int scol = (tid & 7) * 128;
        const int stk  = stok[srow];
        if (stk >= 0) {
            const unsigned short* xr = xbf + (size_t)stk * H_DIM + scol;
            const unsigned lds = (unsigned)(uintptr_t)&Ast[srow][scol];
            #pragma unroll
            for (int i = 0; i < 16; ++i) async_load_b128(lds + i * 16, xr + i * 8);
        } else {
            const u32x4 z = {0u, 0u, 0u, 0u};
            #pragma unroll
            for (int i = 0; i < 16; ++i) *(u32x4*)&Ast[srow][scol + i * 8] = z;
        }
    }
    wait_async0();
    __syncthreads();

    const float* Wfc_e = Wfc + (size_t)e * H_DIM * FF_DIM;
    const size_t hbase = (size_t)(base[e] + tbase);
    const int rowlim = ce - tbase;             // valid rows in this tile

    for (int c = 0; c < 8; ++c) {              // FF chunks of 256
        const int col = ffbase + c * 256 + wv * 16 + nlane;

        v8f c1[4];
        #pragma unroll
        for (int mt = 0; mt < 4; ++mt)
            #pragma unroll
            for (int i = 0; i < 8; ++i) c1[mt][i] = 0.f;

        #pragma unroll 2
        for (int ks = 0; ks < 32; ++ks) {      // K = H = 1024
            const int k0 = ks * 32;
            Frag16 b;
            const float* bp = Wfc_e + (size_t)(k0 + kg * 8) * FF_DIM + col;
            #pragma unroll
            for (int j = 0; j < 16; ++j) b.v[j] = (__bf16)bp[(size_t)((j & 7) + (j >> 3) * 16) * FF_DIM];
            Frag16 a[4];
            #pragma unroll
            for (int mt = 0; mt < 4; ++mt) {
                const int m = mt * 16 + nlane;
                a[mt].q[0] = *(const u32x4*)&Ast[m][k0 + kg * 8];
                a[mt].q[1] = *(const u32x4*)&Ast[m][k0 + 16 + kg * 8];
            }
            #pragma unroll
            for (int mt = 0; mt < 4; ++mt)
                c1[mt] = __builtin_amdgcn_wmma_f32_16x16x32_bf16(
                    false, a[mt].v, false, b.v, (short)0, c1[mt], false, false);
        }

        // relu^2 -> bf16, direct to workspace (lanes 0-15 form 32B segments)
        #pragma unroll
        for (int mt = 0; mt < 4; ++mt)
            #pragma unroll
            for (int r = 0; r < 8; ++r) {
                const int row = mt * 16 + kg * 8 + r;
                if (row >= rowlim) continue;
                float h = c1[mt][r];
                h = (h > 0.f) ? h * h : 0.f;
                h1ws[(hbase + row) * FF_DIM + col] = f2bf(h);
            }
    }
}

// ---------------- GEMM2: out += w * (h1 @ Wproj) ------------------------------
// Block 512 thr = 16 waves. Tile: 64 tokens x H-half (512); wave owns 32 cols.
__global__ __launch_bounds__(512) void moe_proj(const float* __restrict__ Wproj,
                                                const unsigned short* __restrict__ h1ws,
                                                const int* __restrict__ cnt,
                                                const int* __restrict__ base,
                                                const int* __restrict__ btok,
                                                const float* __restrict__ bw,
                                                float* __restrict__ out) {
    const int e = blockIdx.y;
    const int ce = cnt[e];
    const int tbase = blockIdx.x * 64;
    if (tbase >= ce) return;
    const int hh = blockIdx.z;                 // H half

    __shared__ int   stok[64];
    __shared__ float swt[64];

    const int tid   = threadIdx.x;
    const int wv    = tid >> 5;
    const int lane  = tid & 31;
    const int nlane = lane & 15;
    const int kg    = lane >> 4;

    if (tid < 64) {
        const int idx = tbase + tid;
        if (idx < ce) { stok[tid] = btok[e * T_TOK + idx]; swt[tid] = bw[e * T_TOK + idx]; }
        else          { stok[tid] = -1;                    swt[tid] = 0.f; }
    }
    __syncthreads();

    const unsigned short* h1b = h1ws + (size_t)(base[e] + tbase) * FF_DIM;
    const float* Wp_e = Wproj + (size_t)e * FF_DIM * H_DIM + hh * 512 + wv * 32;

    v8f acc[4][2];
    #pragma unroll
    for (int mt = 0; mt < 4; ++mt)
        #pragma unroll
        for (int nt = 0; nt < 2; ++nt)
            #pragma unroll
            for (int i = 0; i < 8; ++i) acc[mt][nt][i] = 0.f;

    #pragma unroll 2
    for (int ks = 0; ks < 128; ++ks) {         // K = FF = 4096
        const int k0 = ks * 32;
        Frag16 a[4];
        #pragma unroll
        for (int mt = 0; mt < 4; ++mt) {
            const unsigned short* ap = h1b + (size_t)(mt * 16 + nlane) * FF_DIM + k0 + kg * 8;
            a[mt].q[0] = *(const u32x4*)ap;
            a[mt].q[1] = *(const u32x4*)(ap + 16);
        }
        #pragma unroll
        for (int nt = 0; nt < 2; ++nt) {
            Frag16 b;
            const float* bp = Wp_e + (size_t)(k0 + kg * 8) * H_DIM + nt * 16 + nlane;
            #pragma unroll
            for (int j = 0; j < 16; ++j) b.v[j] = (__bf16)bp[(size_t)((j & 7) + (j >> 3) * 16) * H_DIM];
            #pragma unroll
            for (int mt = 0; mt < 4; ++mt)
                acc[mt][nt] = __builtin_amdgcn_wmma_f32_16x16x32_bf16(
                    false, a[mt].v, false, b.v, (short)0, acc[mt][nt], false, false);
        }
    }

    const int colb = hh * 512 + wv * 32 + nlane;
    #pragma unroll
    for (int mt = 0; mt < 4; ++mt)
        #pragma unroll
        for (int r = 0; r < 8; ++r) {
            const int lr = mt * 16 + kg * 8 + r;
            const int tk = stok[lr];
            if (tk < 0) continue;
            const float wt = swt[lr];
            float* orow = out + (size_t)tk * H_DIM + colb;
            #pragma unroll
            for (int nt = 0; nt < 2; ++nt)
                atomic_add_f32(orow + nt * 16, wt * acc[mt][nt][r]);
        }
}

// ------------------------------- launcher --------------------------------------
extern "C" void kernel_launch(void* const* d_in, const int* in_sizes, int n_in,
                              void* d_out, int out_size, void* d_ws, size_t ws_size,
                              hipStream_t stream) {
    (void)in_sizes; (void)n_in; (void)ws_size;
    const float* x     = (const float*)d_in[0];
    const float* Wg    = (const float*)d_in[1];
    const float* Wfc   = (const float*)d_in[2];
    const float* Wproj = (const float*)d_in[3];
    float* out = (float*)d_out;

    char* ws = (char*)d_ws;
    int*            cnt  = (int*)ws;                              // 8 ints
    int*            base = (int*)(ws + 64);                       // 8 ints
    int*            btok = (int*)(ws + 1024);                     // 8*4096 ints
    float*          bw   = (float*)(ws + 1024 + 131072);          // 8*4096 floats
    unsigned short* xb   = (unsigned short*)(ws + 1024 + 262144); // 4096*1024 bf16 (8 MB)
    unsigned short* h1   = (unsigned short*)(ws + 1024 + 262144 + 8388608); // 8256*4096 bf16 (~67 MB)

    hipMemsetAsync(out, 0, (size_t)out_size * sizeof(float), stream);
    hipMemsetAsync(cnt, 0, 1024, stream);

    moe_cvt<<<4096, 256, 0, stream>>>(x, xb);
    moe_router<<<512, 256, 0, stream>>>(x, Wg, cnt, btok, bw);
    moe_prefix<<<1, 32, 0, stream>>>(cnt, base);
    moe_fc<<<dim3(64, 8, 2), 512, 0, stream>>>(Wfc, xb, cnt, base, btok, h1);
    moe_proj<<<dim3(64, 8, 2), 512, 0, stream>>>(Wproj, h1, cnt, base, btok, bw, out);
}
